// MambaBackbone_59622736003623
// MI455X (gfx1250) — matrix operand order repaired and checked by reading, not compiled, not run on hardware
//
#include <hip/hip_runtime.h>
#include <hip/hip_bf16.h>
#include <math.h>

// ---------------------------------------------------------------------------
// Mamba backbone forward for gfx1250 (MI455X).
//
// All four GEMMs per layer run on the WMMA pipe (v_wmma_f32_16x16x32_bf16)
// with a bf16 hi/lo split (3 WMMAs per K-step per tile, fp32 accumulate)
// for ~fp32 accuracy.  Operands are pre-packed once per GEMM into
// "fragment-ready" global buffers (one 16x32 WMMA fragment = 32 lanes x 8
// contiguous dwords), so the GEMM inner loop is pure global_load_b128 ->
// v_wmma with no LDS staging and no register shuffling; the 192MB L2 serves
// the re-reads.  Each wave register-blocks a 32x32 output (2x2 tiles -> 12
// WMMAs per 16 b128 loads).  global_prefetch covers the next K-chunk.
// Selective scan maps one state element per lane (wave32, 16 lanes/channel).
// ---------------------------------------------------------------------------

#define D_MODEL_  1024
#define D_INNER_  2048
#define D_STATE_  16
#define D_CONV_   4
#define DT_RANK_  64
#define LSEQ_     2048
#define NLAYERS_  2
#define DBL_W     (DT_RANK_ + 2 * D_STATE_)   // 96
#define EPSF      1e-5f

typedef __attribute__((ext_vector_type(16))) __bf16 v16bf;
typedef __attribute__((ext_vector_type(8)))  float  v8f;

union FragBF { unsigned int u[8]; uint4 q[2]; v16bf v; };

__device__ __forceinline__ unsigned short f32_to_bf16_rne(float f) {
  unsigned int u = __float_as_uint(f);
  u += 0x7fffu + ((u >> 16) & 1u);
  return (unsigned short)(u >> 16);
}
__device__ __forceinline__ float bf16_to_f32(unsigned short h) {
  return __uint_as_float((unsigned int)h << 16);
}
__device__ __forceinline__ float silu_f(float x) {
  return x / (1.0f + __expf(-x));
}

// ---------------------------------------------------------------------------
// Pack a fp32 matrix [R,K] (row stride lda) into WMMA-fragment-ready bf16
// hi/lo buffers.  Tile (g,kc) = rows [16g,16g+16) x K [32kc,32kc+32), stored
// as 256 contiguous dwords: dword (lane*8 + v) holds the bf16 pair for
// K = {pair base} of row (lane&15), per the gfx1250 16-bit A/B layouts:
//   A operand: lanes 0-15 -> K {0..7,16..23}; lanes 16-31 -> K {8..15,24..31}
//   B operand: lanes 0-15 -> K {0..7,16..23}; lanes 16-31 -> K {8..15,24..31}
// Requires R%16==0, K%32==0, Ktiles a power of two (all call sites comply).
// ---------------------------------------------------------------------------
template <bool IS_B>
__global__ __launch_bounds__(256)
void pack_frag_k(const float* __restrict__ src, int lda, int ktLog,
                 unsigned int* __restrict__ hiOut,
                 unsigned int* __restrict__ loOut)
{
  const int idx  = blockIdx.x * 256 + threadIdx.x;   // one bf16-pair each
  const int c    = (idx & 15) << 1;                  // k-in-chunk (even)
  const int m    = (idx >> 4) & 15;                  // row-in-tile
  const int tile = idx >> 8;
  const int kc   = tile & ((1 << ktLog) - 1);
  const int g    = tile >> ktLog;

  const float* p = src + (size_t)(g * 16 + m) * lda + kc * 32 + c;
  const float f0 = p[0], f1 = p[1];
  const unsigned short h0 = f32_to_bf16_rne(f0);
  const unsigned short h1 = f32_to_bf16_rne(f1);
  const unsigned short l0 = f32_to_bf16_rne(f0 - bf16_to_f32(h0));
  const unsigned short l1 = f32_to_bf16_rne(f1 - bf16_to_f32(h1));

  int half, v;
  if (IS_B) {
    half = (c >> 3) & 1;
    v    = ((c & 7) >> 1) + 4 * (c >> 4);
  } else {
    half = (c >> 3) & 1;
    const int cc = c - 8 * half;
    v = (cc < 8) ? (cc >> 1) : (4 + ((cc - 16) >> 1));
  }
  const size_t o = (size_t)tile * 256 + (half * 16 + m) * 8 + v;
  hiOut[o] = (unsigned int)h0 | ((unsigned int)h1 << 16);
  loOut[o] = (unsigned int)l0 | ((unsigned int)l1 << 16);
}

// ---------------------------------------------------------------------------
// WMMA GEMM on packed fragments:  C[M,N] = A[M,K] * W[N,K]^T (+ Res).
// Block: 8 waves as 2(M) x 4(N); each wave computes 2x2 16x16 tiles (32x32).
// Block tile = 64(M) x 128(N).  Per K-step per wave: 16 global b128 loads
// feed 12 WMMAs.  Wave-uniform guards keep EXEC all-ones around WMMA.
// ---------------------------------------------------------------------------
__global__ __launch_bounds__(256)
void gemm_frag(const uint4* __restrict__ Ahi, const uint4* __restrict__ Alo,
               const uint4* __restrict__ Bhi, const uint4* __restrict__ Blo,
               const float* __restrict__ Res, float* __restrict__ C,
               int Ntiles, int Ktiles, int N)
{
  const int tid  = threadIdx.x;
  const int lane = tid & 31;
  const int l16  = lane & 15;
  const int half = lane >> 4;
  const int wave = tid >> 5;

  const int mt0 = (blockIdx.y * 4 + (wave >> 2) * 2);  // first of 2 M-tiles
  const int nt0 = (blockIdx.x * 8 + (wave & 3) * 2);   // first of 2 N-tiles
  const bool n0ok = nt0 < Ntiles;
  const bool n1ok = (nt0 + 1) < Ntiles;

  v8f acc00 = {0.f,0.f,0.f,0.f,0.f,0.f,0.f,0.f};
  v8f acc01 = acc00, acc10 = acc00, acc11 = acc00;

  // uint4 index of this lane's fragment slot inside a 64-uint4 tile.
  const int laneQ = lane * 2;

  size_t a0 = ((size_t)mt0       * Ktiles) * 64 + laneQ;
  size_t a1 = ((size_t)(mt0 + 1) * Ktiles) * 64 + laneQ;
  size_t b0 = ((size_t)nt0       * Ktiles) * 64 + laneQ;
  size_t b1 = ((size_t)(nt0 + 1) * Ktiles) * 64 + laneQ;

  for (int kc = 0; kc < Ktiles; ++kc, a0 += 64, a1 += 64, b0 += 64, b1 += 64) {
    __builtin_prefetch(&Ahi[a0 + 64], 0, 1);
    __builtin_prefetch(&Alo[a0 + 64], 0, 1);
    __builtin_prefetch(&Bhi[b0 + 64], 0, 1);
    __builtin_prefetch(&Blo[b0 + 64], 0, 1);

    FragBF a0h, a0l, a1h, a1l, b0h, b0l, b1h, b1l;
    a0h.q[0] = Ahi[a0]; a0h.q[1] = Ahi[a0 + 1];
    a0l.q[0] = Alo[a0]; a0l.q[1] = Alo[a0 + 1];
    a1h.q[0] = Ahi[a1]; a1h.q[1] = Ahi[a1 + 1];
    a1l.q[0] = Alo[a1]; a1l.q[1] = Alo[a1 + 1];
    if (n0ok) {
      b0h.q[0] = Bhi[b0]; b0h.q[1] = Bhi[b0 + 1];
      b0l.q[0] = Blo[b0]; b0l.q[1] = Blo[b0 + 1];
      // bf16x3: hi*hi + hi*lo + lo*hi
      acc00 = __builtin_amdgcn_wmma_f32_16x16x32_bf16(false, a0h.v, false, b0h.v, (short)0, acc00, false, false);
      acc00 = __builtin_amdgcn_wmma_f32_16x16x32_bf16(false, a0h.v, false, b0l.v, (short)0, acc00, false, false);
      acc00 = __builtin_amdgcn_wmma_f32_16x16x32_bf16(false, a0l.v, false, b0h.v, (short)0, acc00, false, false);
      acc10 = __builtin_amdgcn_wmma_f32_16x16x32_bf16(false, a1h.v, false, b0h.v, (short)0, acc10, false, false);
      acc10 = __builtin_amdgcn_wmma_f32_16x16x32_bf16(false, a1h.v, false, b0l.v, (short)0, acc10, false, false);
      acc10 = __builtin_amdgcn_wmma_f32_16x16x32_bf16(false, a1l.v, false, b0h.v, (short)0, acc10, false, false);
    }
    if (n1ok) {
      b1h.q[0] = Bhi[b1]; b1h.q[1] = Bhi[b1 + 1];
      b1l.q[0] = Blo[b1]; b1l.q[1] = Blo[b1 + 1];
      acc01 = __builtin_amdgcn_wmma_f32_16x16x32_bf16(false, a0h.v, false, b1h.v, (short)0, acc01, false, false);
      acc01 = __builtin_amdgcn_wmma_f32_16x16x32_bf16(false, a0h.v, false, b1l.v, (short)0, acc01, false, false);
      acc01 = __builtin_amdgcn_wmma_f32_16x16x32_bf16(false, a0l.v, false, b1h.v, (short)0, acc01, false, false);
      acc11 = __builtin_amdgcn_wmma_f32_16x16x32_bf16(false, a1h.v, false, b1h.v, (short)0, acc11, false, false);
      acc11 = __builtin_amdgcn_wmma_f32_16x16x32_bf16(false, a1h.v, false, b1l.v, (short)0, acc11, false, false);
      acc11 = __builtin_amdgcn_wmma_f32_16x16x32_bf16(false, a1l.v, false, b1h.v, (short)0, acc11, false, false);
    }
  }

  // C/D layout: lane<16 -> N=l16, VGPR r = M r; lane>=16 -> N=l16, M = 8+r.
  auto store_tile = [&](const v8f& acc, int mt, int nt) {
    const int col = nt * 16 + l16;
    if (nt < Ntiles && col < N) {
      const int rbase = mt * 16 + (half << 3);
#pragma unroll
      for (int r = 0; r < 8; ++r) {
        const size_t o = (size_t)(rbase + r) * N + col;
        float vv = acc[r];
        if (Res) vv += Res[o];
        C[o] = vv;
      }
    }
  };
  store_tile(acc00, mt0,     nt0);
  store_tile(acc10, mt0 + 1, nt0);
  store_tile(acc01, mt0,     nt0 + 1);
  store_tile(acc11, mt0 + 1, nt0 + 1);
}

// ---------------------------------------------------------------------------
// LayerNorm: one block (256 threads = 8 waves) per token, D_MODEL=1024.
// ---------------------------------------------------------------------------
__global__ __launch_bounds__(256)
void layernorm_k(const float* __restrict__ x, const float* __restrict__ w,
                 const float* __restrict__ b, float* __restrict__ out)
{
  __shared__ float s1[8], s2[8];
  const int t = blockIdx.x;
  const float* xr = x + (size_t)t * D_MODEL_;
  float sum = 0.f, sq = 0.f;
  for (int i = threadIdx.x; i < D_MODEL_; i += 256) {
    float v = xr[i];
    sum += v; sq += v * v;
  }
#pragma unroll
  for (int o = 16; o > 0; o >>= 1) {
    sum += __shfl_xor(sum, o, 32);
    sq  += __shfl_xor(sq,  o, 32);
  }
  const int wave = threadIdx.x >> 5, lane = threadIdx.x & 31;
  if (lane == 0) { s1[wave] = sum; s2[wave] = sq; }
  __syncthreads();
  if (wave == 0) {
    sum = (lane < 8) ? s1[lane] : 0.f;
    sq  = (lane < 8) ? s2[lane] : 0.f;
#pragma unroll
    for (int o = 4; o > 0; o >>= 1) {
      sum += __shfl_xor(sum, o, 32);
      sq  += __shfl_xor(sq,  o, 32);
    }
    if (lane == 0) { s1[0] = sum; s2[0] = sq; }
  }
  __syncthreads();
  const float mu  = s1[0] * (1.0f / D_MODEL_);
  const float var = s2[0] * (1.0f / D_MODEL_) - mu * mu;
  const float rs  = rsqrtf(var + EPSF);
  for (int i = threadIdx.x; i < D_MODEL_; i += 256)
    out[(size_t)t * D_MODEL_ + i] = (xr[i] - mu) * rs * w[i] + b[i];
}

// ---------------------------------------------------------------------------
// Causal depthwise conv1d (kernel 4) + SiLU.  Reads x-half of xz[L,2*E].
// ---------------------------------------------------------------------------
__global__ __launch_bounds__(256)
void conv_silu_k(const float* __restrict__ xz, const float* __restrict__ cw,
                 const float* __restrict__ cb, float* __restrict__ xc)
{
  const int idx = blockIdx.x * 256 + threadIdx.x;     // l*E + e
  const int l = idx >> 11;
  const int e = idx & (D_INNER_ - 1);
  float acc = cb[e];
#pragma unroll
  for (int k = 0; k < D_CONV_; ++k) {
    int lp = l - (D_CONV_ - 1) + k;
    if (lp >= 0)
      acc += xz[(size_t)lp * (2 * D_INNER_) + e] * cw[e * D_CONV_ + k];
  }
  xc[idx] = silu_f(acc);
}

// ---------------------------------------------------------------------------
// dt = softplus(dt_gemm + bias)  (in place)
// ---------------------------------------------------------------------------
__global__ __launch_bounds__(256)
void dt_softplus_k(const float* __restrict__ dtg, const float* __restrict__ bias,
                   float* __restrict__ dt)
{
  const int idx = blockIdx.x * 256 + threadIdx.x;
  const int e = idx & (D_INNER_ - 1);
  float v = dtg[idx] + bias[e];
  dt[idx] = (v > 20.0f) ? v : log1pf(__expf(v));
}

// ---------------------------------------------------------------------------
// Selective scan: lane n of each 16-lane group holds state h[e][n]; 2 channels
// per wave (wave32).  Folds the D-skip and silu(z) gate into the write.
// ---------------------------------------------------------------------------
__global__ __launch_bounds__(256)
void scan_k(const float* __restrict__ dt, const float* __restrict__ xc,
            const float* __restrict__ dbl, const float* __restrict__ xz,
            const float* __restrict__ A_log, const float* __restrict__ Dp,
            float* __restrict__ y)
{
  const int n   = threadIdx.x & 15;
  const int grp = threadIdx.x >> 4;
  const int e   = blockIdx.x * 16 + grp;
  const float A  = -__expf(A_log[e * D_STATE_ + n]);
  const float De = Dp[e];
  float h = 0.f;
  for (int l = 0; l < LSEQ_; ++l) {
    const float dtv = dt[(size_t)l * D_INNER_ + e];
    const float xv  = xc[(size_t)l * D_INNER_ + e];
    const float Bv  = dbl[(size_t)l * DBL_W + DT_RANK_ + n];
    const float Cv  = dbl[(size_t)l * DBL_W + DT_RANK_ + D_STATE_ + n];
    h = __expf(dtv * A) * h + (dtv * xv) * Bv;
    float p = h * Cv;
    p += __shfl_xor(p, 1, 32);
    p += __shfl_xor(p, 2, 32);
    p += __shfl_xor(p, 4, 32);
    p += __shfl_xor(p, 8, 32);                      // stays within 16-group
    if (n == 0) {
      const float zv = xz[(size_t)l * (2 * D_INNER_) + D_INNER_ + e];
      y[(size_t)l * D_INNER_ + e] = (p + xv * De) * silu_f(zv);
    }
  }
}

__global__ __launch_bounds__(256)
void copy_k(const float* __restrict__ src, float* __restrict__ dst, int n)
{
  int i = blockIdx.x * 256 + threadIdx.x;
  if (i < n) dst[i] = src[i];
}

// ---------------------------------------------------------------------------
// Workspace layout (in 4-byte units)
// ---------------------------------------------------------------------------
static const size_t OFF_XCUR = 0;
static const size_t OFF_NORM = OFF_XCUR + (size_t)LSEQ_ * D_MODEL_;
static const size_t OFF_XZ   = OFF_NORM + (size_t)LSEQ_ * D_MODEL_;
static const size_t OFF_XC   = OFF_XZ   + (size_t)LSEQ_ * 2 * D_INNER_;
static const size_t OFF_DBL  = OFF_XC   + (size_t)LSEQ_ * D_INNER_;
static const size_t OFF_DT   = OFF_DBL  + (size_t)LSEQ_ * DBL_W;
static const size_t OFF_Y    = OFF_DT   + (size_t)LSEQ_ * D_INNER_;
// packed bf16 fragment buffers (dwords): max A = L*2048/2, max W = 4096*1024/2
static const size_t PK_A     = (size_t)LSEQ_ * D_INNER_ / 2;
static const size_t PK_W     = (size_t)(2 * D_INNER_) * D_MODEL_ / 2;
static const size_t OFF_PAH  = OFF_Y   + (size_t)LSEQ_ * D_INNER_;
static const size_t OFF_PAL  = OFF_PAH + PK_A;
static const size_t OFF_PWH  = OFF_PAL + PK_A;
static const size_t OFF_PWL  = OFF_PWH + PK_W;

extern "C" void kernel_launch(void* const* d_in, const int* in_sizes, int n_in,
                              void* d_out, int out_size, void* d_ws, size_t ws_size,
                              hipStream_t stream)
{
  (void)in_sizes; (void)n_in; (void)out_size; (void)ws_size;

  const float* x_in   = (const float*)d_in[0];
  const float* ln_w   = (const float*)d_in[1];
  const float* ln_b   = (const float*)d_in[2];
  const float* in_w   = (const float*)d_in[3];
  const float* conv_w = (const float*)d_in[4];
  const float* conv_b = (const float*)d_in[5];
  const float* xp_w   = (const float*)d_in[6];
  const float* dtp_w  = (const float*)d_in[7];
  const float* dtp_b  = (const float*)d_in[8];
  const float* A_log  = (const float*)d_in[9];
  const float* D_skip = (const float*)d_in[10];
  const float* out_w  = (const float*)d_in[11];
  const float* nf_w   = (const float*)d_in[12];
  const float* nf_b   = (const float*)d_in[13];
  float* out = (float*)d_out;

  float* ws    = (float*)d_ws;
  float* x_cur = ws + OFF_XCUR;
  float* normd = ws + OFF_NORM;
  float* xz    = ws + OFF_XZ;
  float* xc    = ws + OFF_XC;
  float* dbl   = ws + OFF_DBL;
  float* dtb   = ws + OFF_DT;
  float* yb    = ws + OFF_Y;
  unsigned int* pAh = (unsigned int*)(ws + OFF_PAH);
  unsigned int* pAl = (unsigned int*)(ws + OFF_PAL);
  unsigned int* pWh = (unsigned int*)(ws + OFF_PWH);
  unsigned int* pWl = (unsigned int*)(ws + OFF_PWL);

  copy_k<<<(LSEQ_ * D_MODEL_) / 256, 256, 0, stream>>>(x_in, x_cur, LSEQ_ * D_MODEL_);

  const int MT = LSEQ_ / 16;     // 128 M-tiles -> grid.y = 32

  for (int i = 0; i < NLAYERS_; ++i) {
    // ---- LN ----
    layernorm_k<<<LSEQ_, 256, 0, stream>>>(x_cur, ln_w + i * D_MODEL_,
                                           ln_b + i * D_MODEL_, normd);

    // ---- in_proj: [L,1024] x [4096,1024]^T -> xz[L,4096]  (K=1024, kt=32) ----
    pack_frag_k<false><<<(LSEQ_ * D_MODEL_ / 2) / 256, 256, 0, stream>>>(
        normd, D_MODEL_, 5, pAh, pAl);
    pack_frag_k<true><<<((2 * D_INNER_) * D_MODEL_ / 2) / 256, 256, 0, stream>>>(
        in_w + (size_t)i * 2 * D_INNER_ * D_MODEL_, D_MODEL_, 5, pWh, pWl);
    gemm_frag<<<dim3((2 * D_INNER_ / 16 + 7) / 8, MT / 4), 256, 0, stream>>>(
        (const uint4*)pAh, (const uint4*)pAl, (const uint4*)pWh, (const uint4*)pWl,
        nullptr, xz, 2 * D_INNER_ / 16, D_MODEL_ / 32, 2 * D_INNER_);

    // ---- causal conv + SiLU ----
    conv_silu_k<<<(LSEQ_ * D_INNER_) / 256, 256, 0, stream>>>(
        xz, conv_w + (size_t)i * D_INNER_ * D_CONV_, conv_b + i * D_INNER_, xc);

    // ---- x_proj: [L,2048] x [96,2048]^T -> dbl[L,96]  (K=2048, kt=64) ----
    pack_frag_k<false><<<(LSEQ_ * D_INNER_ / 2) / 256, 256, 0, stream>>>(
        xc, D_INNER_, 6, pAh, pAl);
    pack_frag_k<true><<<(DBL_W * D_INNER_ / 2) / 256, 256, 0, stream>>>(
        xp_w + (size_t)i * DBL_W * D_INNER_, D_INNER_, 6, pWh, pWl);
    gemm_frag<<<dim3((DBL_W / 16 + 7) / 8, MT / 4), 256, 0, stream>>>(
        (const uint4*)pAh, (const uint4*)pAl, (const uint4*)pWh, (const uint4*)pWl,
        nullptr, dbl, DBL_W / 16, D_INNER_ / 32, DBL_W);

    // ---- dt_proj: dt cols of dbl (lda=96,K=64,kt=2) x [2048,64]^T -> dtb ----
    pack_frag_k<false><<<(LSEQ_ * DT_RANK_ / 2) / 256, 256, 0, stream>>>(
        dbl, DBL_W, 1, pAh, pAl);
    pack_frag_k<true><<<(D_INNER_ * DT_RANK_ / 2) / 256, 256, 0, stream>>>(
        dtp_w + (size_t)i * D_INNER_ * DT_RANK_, DT_RANK_, 1, pWh, pWl);
    gemm_frag<<<dim3((D_INNER_ / 16 + 7) / 8, MT / 4), 256, 0, stream>>>(
        (const uint4*)pAh, (const uint4*)pAl, (const uint4*)pWh, (const uint4*)pWl,
        nullptr, dtb, D_INNER_ / 16, DT_RANK_ / 32, D_INNER_);

    // ---- softplus(+bias) in place ----
    dt_softplus_k<<<(LSEQ_ * D_INNER_) / 256, 256, 0, stream>>>(
        dtb, dtp_b + i * D_INNER_, dtb);

    // ---- selective scan + D-skip + silu(z) gate -> yb ----
    scan_k<<<D_INNER_ / 16, 256, 0, stream>>>(
        dtb, xc, dbl, xz, A_log + (size_t)i * D_INNER_ * D_STATE_,
        D_skip + i * D_INNER_, yb);

    // ---- out_proj + fused residual: x_cur = yb x [1024,2048]^T + x_cur ----
    pack_frag_k<false><<<(LSEQ_ * D_INNER_ / 2) / 256, 256, 0, stream>>>(
        yb, D_INNER_, 6, pAh, pAl);
    pack_frag_k<true><<<(D_MODEL_ * D_INNER_ / 2) / 256, 256, 0, stream>>>(
        out_w + (size_t)i * D_MODEL_ * D_INNER_, D_INNER_, 6, pWh, pWl);
    gemm_frag<<<dim3((D_MODEL_ / 16 + 7) / 8, MT / 4), 256, 0, stream>>>(
        (const uint4*)pAh, (const uint4*)pAl, (const uint4*)pWh, (const uint4*)pWl,
        x_cur, x_cur, D_MODEL_ / 16, D_INNER_ / 32, D_MODEL_);
  }

  // ---- final LN -> d_out ----
  layernorm_k<<<LSEQ_, 256, 0, stream>>>(x_cur, nf_w, nf_b, out);
}
